// GNNWithFA_82240033784023
// MI455X (gfx1250) — matrix operand order repaired and checked by reading, not compiled, not run on hardware
//
#include <hip/hip_runtime.h>
#include <hip/hip_bf16.h>

typedef _Float16 h8  __attribute__((ext_vector_type(8)));
typedef _Float16 h16 __attribute__((ext_vector_type(16)));
typedef float    f8  __attribute__((ext_vector_type(8)));
typedef float    f4  __attribute__((ext_vector_type(4)));

#define NNODES   16384
#define DFEAT    128
#define NEDGES   524288
#define NPERG    1024
#define NHEAD    4
#define HDIM     32
#define OUTD     64
#define BN_SCALE 0.9999950000374997f   // 1/sqrt(1+1e-5)
#define LN_EPS   1e-5f

// ---------------------------------------------------------------- helpers
__device__ __forceinline__ h16 cat8(h8 a, h8 b) {
  return __builtin_shufflevector(a, b, 0,1,2,3,4,5,6,7,8,9,10,11,12,13,14,15);
}
__device__ __forceinline__ f8 wmma16(h16 a, h16 b, f8 c) {
  return __builtin_amdgcn_wmma_f32_16x16x32_f16(false, a, false, b, (short)0, c, false, false);
}
__device__ __forceinline__ float redmax16(float v) {
#pragma unroll
  for (int m = 1; m < 16; m <<= 1) v = fmaxf(v, __shfl_xor(v, m, 32));
  return v;
}
__device__ __forceinline__ float redsum16(float v) {
#pragma unroll
  for (int m = 1; m < 16; m <<= 1) v += __shfl_xor(v, m, 32);
  return v;
}
__device__ __forceinline__ float redsum32(float v) {
#pragma unroll
  for (int m = 1; m < 32; m <<= 1) v += __shfl_xor(v, m, 32);
  return v;
}

// Load one wave's A fragments (16 rows x 128 K, fp32 -> f16) for 4 k-chunks.
// A layout (16-bit, 16x32): lanes 0-15 hold K=32c+{0..7,16..23}, lanes 16-31
// hold K=32c+{8..15,24..31}, row M = lane%16.
__device__ __forceinline__ void load_A_f32(const float* __restrict__ X, int rowBase,
                                           int lane, h16 A[4]) {
  const int l16 = lane & 15;
  const int s   = (lane & 16) ? 8 : 0;
  const float* xr = X + (size_t)(rowBase + l16) * DFEAT;
#pragma unroll
  for (int c = 0; c < 4; ++c) {
    const float* p = xr + 32 * c + s;
    f4 a0 = *(const f4*)(p);
    f4 a1 = *(const f4*)(p + 4);
    f4 a2 = *(const f4*)(p + 16);
    f4 a3 = *(const f4*)(p + 20);
    h16 t;
#pragma unroll
    for (int j = 0; j < 4; ++j) {
      t[j]      = (_Float16)a0[j];
      t[4 + j]  = (_Float16)a1[j];
      t[8 + j]  = (_Float16)a2[j];
      t[12 + j] = (_Float16)a3[j];
    }
    A[c] = t;
  }
}

// Load 4 B-fragment chunks (K = 0..127) for one 16-wide n-tile.
__device__ __forceinline__ void load_B4(const _Float16* __restrict__ wr, h16 B[4]) {
#pragma unroll
  for (int c = 0; c < 4; ++c)
    B[c] = cat8(*(const h8*)(wr + 32 * c), *(const h8*)(wr + 32 * c + 16));
}

// ---------------------------------------------------------------- weights
// W [K x Nout] fp32 -> WT [Nout x K] f16
__global__ __launch_bounds__(256) void k_wt(const float* __restrict__ W,
                                            _Float16* __restrict__ WT,
                                            int K, int Nout) {
  int i = blockIdx.x * 256 + threadIdx.x;
  if (i >= K * Nout) return;
  int n = i / K, k = i % K;
  WT[n * K + k] = (_Float16)W[k * Nout + n];
}

// Fold GEMM epilogue into per-column alpha/beta:
//   y = (acc + b0)*s*smul + b1  ==  acc*alpha + beta
__global__ __launch_bounds__(128) void k_mkab(const float* __restrict__ b0,
                                              const float* __restrict__ s,
                                              float smul,
                                              const float* __restrict__ b1,
                                              float* __restrict__ ab, int n) {
  int i = threadIdx.x;
  if (i >= n) return;
  float sc = s ? s[i] * smul : 1.f;
  float v0 = b0 ? b0[i] : 0.f;
  float v1 = b1 ? b1[i] : 0.f;
  ab[i]       = sc;
  ab[128 + i] = v0 * sc + v1;
}

// ---------------------------------------------------------------- GIN prep
__global__ __launch_bounds__(256) void k_scaled_copy(const float* __restrict__ h,
                                                     const float* __restrict__ epss,
                                                     int layer, float* __restrict__ out) {
  int i = blockIdx.x * 256 + threadIdx.x;           // over N*D/4
  float e = 1.0f + epss[layer];
  f4 v = *(const f4*)(h + (size_t)i * 4);
  f4 r;
#pragma unroll
  for (int j = 0; j < 4; ++j) r[j] = v[j] * e;
  *(f4*)(out + (size_t)i * 4) = r;
}

__global__ __launch_bounds__(256) void k_edge_scatter(const float* __restrict__ h,
                                                      const long long* __restrict__ ei,
                                                      float* __restrict__ out) {
  int t = blockIdx.x * 256 + threadIdx.x;           // E*32 threads, 4 floats each
  int e = t >> 5;
  int c = (t & 31) * 4;
  int src = (int)ei[e];
  int dst = (int)ei[NEDGES + e];
  f4 v = *(const f4*)(h + (size_t)src * DFEAT + c);
  float* o = out + (size_t)dst * DFEAT + c;
  atomicAdd(o + 0, v[0]);
  atomicAdd(o + 1, v[1]);
  atomicAdd(o + 2, v[2]);
  atomicAdd(o + 3, v[3]);
}

// ---------------------------------------------------------------- GEMM (fp32 out)
// Y[row, col] = maybe_relu(acc * alpha[col] + beta[col]) (+ residual[row, col])
template <int RELU, int RES, int DOUT>
__global__ __launch_bounds__(256) void k_gemm_ep(const float* __restrict__ X,
                                                 const _Float16* __restrict__ WT,
                                                 const float* __restrict__ ab,
                                                 const float* __restrict__ residual,
                                                 float* __restrict__ Y) {
  const int lane = threadIdx.x & 31;
  const int wave = threadIdx.x >> 5;
  const int rowBase = blockIdx.x * 128 + wave * 16;
  const int l16 = lane & 15;
  const int hi  = (lane & 16) ? 1 : 0;
  const int koff = hi ? 8 : 0;

  h16 A[4];
  load_A_f32(X, rowBase, lane, A);

  constexpr int NT = DOUT >> 4;
  h16 Bn[4];
  load_B4(WT + (size_t)l16 * DFEAT + koff, Bn);   // tile 0

  for (int nt = 0; nt < NT; ++nt) {
    h16 B0 = Bn[0], B1 = Bn[1], B2 = Bn[2], B3 = Bn[3];
    const int ntn = (nt + 1 < NT) ? nt + 1 : NT - 1;
    load_B4(WT + (size_t)(ntn * 16 + l16) * DFEAT + koff, Bn);  // prefetch next

    f8 acc = {};
    acc = wmma16(A[0], B0, acc);
    acc = wmma16(A[1], B1, acc);
    acc = wmma16(A[2], B2, acc);
    acc = wmma16(A[3], B3, acc);

    const int col = nt * 16 + l16;
    const float al = ab[col];
    const float be = ab[128 + col];
    float* yp = Y + (size_t)(rowBase + hi * 8) * DOUT + col;
    const float* rp = residual + (size_t)(rowBase + hi * 8) * DOUT + col;
#pragma unroll
    for (int r = 0; r < 8; ++r) {
      float y = acc[r] * al + be;
      if (RELU) y = fmaxf(y, 0.f);
      if (RES)  y += rp[r * DOUT];
      yp[r * DOUT] = y;
    }
  }
}

// ---------------------------------------------------------------- QKV GEMM (f16 out)
// VMODE 0: Y[((b*H+h)*NG + node)*HD + hd]   (Q, K row-major per head)
// VMODE 1: Y[((b*H+h)*HD + hd)*NG + node]   (V transposed per head)
template <int VMODE>
__global__ __launch_bounds__(256) void k_gemm_qkv(const float* __restrict__ X,
                                                  const _Float16* __restrict__ WT,
                                                  const float* __restrict__ bias,
                                                  _Float16* __restrict__ Yf) {
  const int lane = threadIdx.x & 31;
  const int wave = threadIdx.x >> 5;
  const int rowBase = blockIdx.x * 128 + wave * 16;
  const int l16 = lane & 15;
  const int hi  = (lane & 16) ? 1 : 0;
  const int koff = hi ? 8 : 0;

  h16 A[4];
  load_A_f32(X, rowBase, lane, A);

  h16 Bn[4];
  load_B4(WT + (size_t)l16 * DFEAT + koff, Bn);

  for (int nt = 0; nt < 8; ++nt) {
    h16 B0 = Bn[0], B1 = Bn[1], B2 = Bn[2], B3 = Bn[3];
    const int ntn = (nt + 1 < 8) ? nt + 1 : 7;
    load_B4(WT + (size_t)(ntn * 16 + l16) * DFEAT + koff, Bn);

    f8 acc = {};
    acc = wmma16(A[0], B0, acc);
    acc = wmma16(A[1], B1, acc);
    acc = wmma16(A[2], B2, acc);
    acc = wmma16(A[3], B3, acc);

    const int col = nt * 16 + l16;
    const int hh = col >> 5, hd = col & 31;
    const float bv = bias[col];
    const int row0 = rowBase + hi * 8;           // rows row0..row0+7, same graph
    const int b = row0 >> 10, node = row0 & 1023;
    _Float16* yp;
    if (VMODE == 0)
      yp = Yf + ((size_t)((b * NHEAD + hh) * NPERG + node)) * HDIM + hd;
    else
      yp = Yf + ((size_t)((b * NHEAD + hh) * HDIM + hd)) * NPERG + node;
#pragma unroll
    for (int r = 0; r < 8; ++r) {
      float y = acc[r] + bv;
      yp[VMODE == 0 ? r * HDIM : r] = (_Float16)y;
    }
  }
}

// ---------------------------------------------------------------- flash attention
// grid: (B*H)*8 blocks; block = 8 waves; wave = 16 queries vs all 1024 keys.
__device__ __forceinline__ void load_kv(const _Float16* __restrict__ Kb,
                                        const _Float16* __restrict__ Vb,
                                        int key0, int l16, int koff,
                                        h16& Bk0, h16& Bk1, h16& Bv0, h16& Bv1) {
  const _Float16* kr0 = Kb + (size_t)(key0 + l16) * HDIM + koff;
  const _Float16* kr1 = kr0 + 16 * HDIM;
  Bk0 = cat8(*(const h8*)kr0, *(const h8*)(kr0 + 16));
  Bk1 = cat8(*(const h8*)kr1, *(const h8*)(kr1 + 16));
  const _Float16* vr0 = Vb + (size_t)l16 * NPERG + key0 + koff;
  const _Float16* vr1 = vr0 + 16 * NPERG;
  Bv0 = cat8(*(const h8*)vr0, *(const h8*)(vr0 + 16));
  Bv1 = cat8(*(const h8*)vr1, *(const h8*)(vr1 + 16));
}

__global__ __launch_bounds__(256) void k_attn(const _Float16* __restrict__ Qf,
                                              const _Float16* __restrict__ Kf,
                                              const _Float16* __restrict__ Vt,
                                              float* __restrict__ O) {
  __shared__ __align__(16) _Float16 plds[8 * 512];  // wave-private 16x32 P tiles
  const int lane = threadIdx.x & 31;
  const int wave = threadIdx.x >> 5;
  const int l16 = lane & 15;
  const int hi  = (lane & 16) ? 1 : 0;
  const int koff = hi ? 8 : 0;
  const int bh  = blockIdx.x >> 3;
  const int qblk = blockIdx.x & 7;
  const int b = bh >> 2, hh = bh & 3;

  const _Float16* Qb = Qf + (size_t)bh * NPERG * HDIM;
  const _Float16* Kb = Kf + (size_t)bh * NPERG * HDIM;
  const _Float16* Vb = Vt + (size_t)bh * HDIM * NPERG;
  const int qbase = qblk * 128 + wave * 16;

  // Q A-fragment (16 x 32, k = head dim)
  h16 Aq;
  {
    const int s = hi ? 8 : 0;
    const _Float16* qr = Qb + (size_t)(qbase + l16) * HDIM + s;
    Aq = cat8(*(const h8*)qr, *(const h8*)(qr + 16));
  }

  float m[8], lsum[8];
  f8 o0 = {}, o1 = {};
#pragma unroll
  for (int r = 0; r < 8; ++r) { m[r] = -3.0e38f; lsum[r] = 0.f; }

  _Float16* pw = plds + wave * 512;
  const float sm = 0.17677669529663687f;  // 1/sqrt(32)

  h16 nBk0, nBk1, nBv0, nBv1;
  load_kv(Kb, Vb, 0, l16, koff, nBk0, nBk1, nBv0, nBv1);

  for (int kt = 0; kt < 32; ++kt) {
    h16 Bk0 = nBk0, Bk1 = nBk1, Bv0 = nBv0, Bv1 = nBv1;
    const int ktn = (kt + 1) & 31;                      // branch-free prefetch
    load_kv(Kb, Vb, ktn * 32, l16, koff, nBk0, nBk1, nBv0, nBv1);

    // S = Q K^T for 32 keys (two 16-key n-tiles)
    f8 s0 = {}, s1 = {};
    s0 = wmma16(Aq, Bk0, s0);
    s1 = wmma16(Aq, Bk1, s1);
#pragma unroll
    for (int r = 0; r < 8; ++r) { s0[r] *= sm; s1[r] *= sm; }

    // online softmax
#pragma unroll
    for (int r = 0; r < 8; ++r) {
      float rmax = redmax16(fmaxf(s0[r], s1[r]));
      float mn = fmaxf(m[r], rmax);
      float corr = __expf(m[r] - mn);
      m[r] = mn;
      o0[r] *= corr;
      o1[r] *= corr;
      lsum[r] *= corr;
      float p0 = __expf(s0[r] - mn);
      float p1 = __expf(s1[r] - mn);
      s0[r] = p0; s1[r] = p1;
      lsum[r] += redsum16(p0 + p1);
    }

    // transpose P (C-layout -> A-layout) through wave-private LDS
#pragma unroll
    for (int r = 0; r < 8; ++r) {
      int row = r + (hi ? 8 : 0);
      pw[row * 32 + l16]      = (_Float16)s0[r];
      pw[row * 32 + 16 + l16] = (_Float16)s1[r];
    }
    asm volatile("s_wait_dscnt 0x0" ::: "memory");
    const int s = hi ? 8 : 0;
    const _Float16* pr = pw + l16 * 32 + s;
    h16 Ap = cat8(*(const h8*)pr, *(const h8*)(pr + 16));

    // O += P V
    o0 = wmma16(Ap, Bv0, o0);
    o1 = wmma16(Ap, Bv1, o1);
  }

  float* op = O + ((size_t)b * NPERG + qbase + hi * 8) * DFEAT + hh * HDIM + l16;
#pragma unroll
  for (int r = 0; r < 8; ++r) {
    float inv = 1.f / lsum[r];
    op[r * DFEAT]      = o0[r] * inv;
    op[r * DFEAT + 16] = o1[r] * inv;
  }
}

// ---------------------------------------------------------------- layernorm
__global__ __launch_bounds__(256) void k_layernorm(const float* __restrict__ X,
                                                   const float* __restrict__ g,
                                                   const float* __restrict__ bta,
                                                   float* __restrict__ Y) {
  const int lane = threadIdx.x & 31;
  const int wave = threadIdx.x >> 5;
  const int row = blockIdx.x * 8 + wave;
  const float* xr = X + (size_t)row * DFEAT;
  f4 x = *(const f4*)(xr + lane * 4);
  float s = x[0] + x[1] + x[2] + x[3];
  float s2 = x[0] * x[0] + x[1] * x[1] + x[2] * x[2] + x[3] * x[3];
  s = redsum32(s);
  s2 = redsum32(s2);
  float mean = s * (1.f / DFEAT);
  float var = s2 * (1.f / DFEAT) - mean * mean;
  float inv = rsqrtf(var + LN_EPS);
  f4 y;
#pragma unroll
  for (int j = 0; j < 4; ++j)
    y[j] = (x[j] - mean) * inv * g[lane * 4 + j] + bta[lane * 4 + j];
  *(f4*)(Y + (size_t)row * DFEAT + lane * 4) = y;
}

// ---------------------------------------------------------------- launch
extern "C" void kernel_launch(void* const* d_in, const int* in_sizes, int n_in,
                              void* d_out, int out_size, void* d_ws, size_t ws_size,
                              hipStream_t stream) {
  (void)in_sizes; (void)n_in; (void)out_size; (void)ws_size;

  const float* x        = (const float*)d_in[0];
  const long long* ei   = (const long long*)d_in[1];
  const float* W1s      = (const float*)d_in[2];
  const float* b1s      = (const float*)d_in[3];
  const float* g1s      = (const float*)d_in[4];
  const float* bt1s     = (const float*)d_in[5];
  const float* W2s      = (const float*)d_in[6];
  const float* b2s      = (const float*)d_in[7];
  const float* epss     = (const float*)d_in[8];
  const float* ogs      = (const float*)d_in[9];
  const float* obs      = (const float*)d_in[10];
  const float* Wq       = (const float*)d_in[11];
  const float* bq       = (const float*)d_in[12];
  const float* Wk       = (const float*)d_in[13];
  const float* bk       = (const float*)d_in[14];
  const float* Wv       = (const float*)d_in[15];
  const float* bv       = (const float*)d_in[16];
  const float* Wo       = (const float*)d_in[17];
  const float* bo       = (const float*)d_in[18];
  const float* ln_g     = (const float*)d_in[19];
  const float* ln_b     = (const float*)d_in[20];
  const float* Wh       = (const float*)d_in[21];
  const float* bh       = (const float*)d_in[22];
  float* out            = (float*)d_out;

  // workspace layout (~44.4 MB total)
  char* ws = (char*)d_ws;
  const size_t F32MAT = (size_t)NNODES * DFEAT * sizeof(float);    // 8 MiB
  const size_t F16MAT = (size_t)NNODES * DFEAT * sizeof(_Float16); // 4 MiB
  float*     hA    = (float*)(ws);
  float*     hB    = (float*)(ws + F32MAT);
  float*     hC    = (float*)(ws + 2 * F32MAT);
  float*     attnO = (float*)(ws + 3 * F32MAT);
  _Float16*  Qf    = (_Float16*)(ws + 4 * F32MAT);
  _Float16*  Kf    = (_Float16*)(ws + 4 * F32MAT + F16MAT);
  _Float16*  Vt    = (_Float16*)(ws + 4 * F32MAT + 2 * F16MAT);
  _Float16*  WT    = (_Float16*)(ws + 4 * F32MAT + 3 * F16MAT);
  const size_t MSLOT = (size_t)DFEAT * DFEAT;  // 16384 halfs
  _Float16* W1T = WT;                 // 3 slots
  _Float16* W2T = WT + 3 * MSLOT;     // 3 slots
  _Float16* WqT = WT + 6 * MSLOT;
  _Float16* WkT = WT + 7 * MSLOT;
  _Float16* WvT = WT + 8 * MSLOT;
  _Float16* WoT = WT + 9 * MSLOT;
  _Float16* WhT = WT + 10 * MSLOT;    // 64*128 halfs
  float*    ab  = (float*)(WT + 11 * MSLOT);   // alpha[128] + beta[128]

  // --- convert + transpose all weights to f16 ---
  for (int i = 0; i < 3; ++i) {
    k_wt<<<64, 256, 0, stream>>>(W1s + i * MSLOT, W1T + i * MSLOT, DFEAT, DFEAT);
    k_wt<<<64, 256, 0, stream>>>(W2s + i * MSLOT, W2T + i * MSLOT, DFEAT, DFEAT);
  }
  k_wt<<<64, 256, 0, stream>>>(Wq, WqT, DFEAT, DFEAT);
  k_wt<<<64, 256, 0, stream>>>(Wk, WkT, DFEAT, DFEAT);
  k_wt<<<64, 256, 0, stream>>>(Wv, WvT, DFEAT, DFEAT);
  k_wt<<<64, 256, 0, stream>>>(Wo, WoT, DFEAT, DFEAT);
  k_wt<<<32, 256, 0, stream>>>(Wh, WhT, DFEAT, OUTD);

  // --- GIN stack ---
  const int gemmBlocks = NNODES / 128;              // 128
  const int elemBlocks = NNODES * DFEAT / 4 / 256;  // 2048
  const int edgeBlocks = NEDGES * 32 / 256;         // 65536
  const float* h = x;
  for (int i = 0; i < 3; ++i) {
    k_scaled_copy<<<elemBlocks, 256, 0, stream>>>(h, epss, i, hB);
    k_edge_scatter<<<edgeBlocks, 256, 0, stream>>>(h, ei, hB);
    // z = relu((hB @ W1 + b1) * BN_SCALE*g1 + bt1)
    k_mkab<<<1, 128, 0, stream>>>(b1s + i * DFEAT, g1s + i * DFEAT, BN_SCALE,
                                  bt1s + i * DFEAT, ab, DFEAT);
    k_gemm_ep<1, 0, DFEAT><<<gemmBlocks, 256, 0, stream>>>(hB, W1T + i * MSLOT,
                                                           ab, nullptr, hC);
    if (i < 2) {
      // h = relu((hC @ W2 + b2) * BN_SCALE*ogs + obs)
      k_mkab<<<1, 128, 0, stream>>>(b2s + i * DFEAT, ogs + i * DFEAT, BN_SCALE,
                                    obs + i * DFEAT, ab, DFEAT);
      k_gemm_ep<1, 0, DFEAT><<<gemmBlocks, 256, 0, stream>>>(hC, W2T + i * MSLOT,
                                                             ab, nullptr, hA);
    } else {
      // h = hC @ W2 + b2
      k_mkab<<<1, 128, 0, stream>>>(b2s + i * DFEAT, nullptr, 1.f, nullptr, ab, DFEAT);
      k_gemm_ep<0, 0, DFEAT><<<gemmBlocks, 256, 0, stream>>>(hC, W2T + i * MSLOT,
                                                             ab, nullptr, hA);
    }
    h = hA;
  }

  // --- FA layer ---
  k_gemm_qkv<0><<<gemmBlocks, 256, 0, stream>>>(hA, WqT, bq, Qf);
  k_gemm_qkv<0><<<gemmBlocks, 256, 0, stream>>>(hA, WkT, bk, Kf);
  k_gemm_qkv<1><<<gemmBlocks, 256, 0, stream>>>(hA, WvT, bv, Vt);
  k_attn<<<16 * NHEAD * 8, 256, 0, stream>>>(Qf, Kf, Vt, attnO);
  // hB = hA + attnO @ Wo + bo
  k_mkab<<<1, 128, 0, stream>>>(bo, nullptr, 1.f, nullptr, ab, DFEAT);
  k_gemm_ep<0, 1, DFEAT><<<gemmBlocks, 256, 0, stream>>>(attnO, WoT, ab, hA, hB);
  // layernorm
  k_layernorm<<<NNODES / 8, 256, 0, stream>>>(hB, ln_g, ln_b, hC);
  // head: out = hC @ Wh + bh  [N, 64]
  k_mkab<<<1, 128, 0, stream>>>(bh, nullptr, 1.f, nullptr, ab, OUTD);
  k_gemm_ep<0, 0, OUTD><<<gemmBlocks, 256, 0, stream>>>(hC, WhT, ab, nullptr, out);
}
